// TWS_55594056680192
// MI455X (gfx1250) — compile-verified, matrix operand order checked by reading, not run on hardware
//
#include <hip/hip_runtime.h>

typedef float        v8f  __attribute__((ext_vector_type(8)));
typedef float        v4f  __attribute__((ext_vector_type(4)));
typedef unsigned int v4u  __attribute__((ext_vector_type(4)));
typedef unsigned int v2u  __attribute__((ext_vector_type(2)));
typedef __bf16       v16bf __attribute__((ext_vector_type(16)));

#define D_DIM   768
#define E_NUM   100
#define MT_ROWS 64                  // rows per block (16 blocks/expert)
#define NRT     (MT_ROWS / 16)      // 4 row-tiles
#define XS_STR  776                 // 768 + 8 halfs pad -> 388 dwords, 388%64=4 (conflict-free b128)
#define WB_STR  72                  // 64 + 8 halfs pad  -> 36 dwords, conflict-free b128
#define NCHUNK  128
#define KCH     64                  // staged K per barrier (2 WMMA k-steps)
#define NKB     (D_DIM / KCH)       // 12
#define NNC     (D_DIM / NCHUNK)    // 6
#define WBUF    (NCHUNK * WB_STR)   // 9216 halfs = 18 KB per stage buffer

// ---- bf16 pack: ONE v_perm_b32 per two floats (truncate to high 16 bits) ----
__device__ __forceinline__ unsigned pk2bf(float lo, float hi) {
  return __builtin_amdgcn_perm(__builtin_bit_cast(unsigned int, hi),
                               __builtin_bit_cast(unsigned int, lo),
                               0x07060302u);
}
__device__ __forceinline__ unsigned short f2bf1(float f) {
  unsigned u = __builtin_bit_cast(unsigned int, f);
  return (unsigned short)((u + 0x8000u) >> 16);
}

// ---- branch-free exact-erf GELU (A&S 7.1.26, |err|<1.5e-7) ----
__device__ __forceinline__ float gelu_exact(float v) {
  float z  = v * 0.70710678118654752f;
  float az = fabsf(z);
  float t  = __builtin_amdgcn_rcpf(fmaf(0.3275911f, az, 1.0f));
  float p  = fmaf(t, 1.061405429f, -1.453152027f);
  p = fmaf(t, p, 1.421413741f);
  p = fmaf(t, p, -0.284496736f);
  p = fmaf(t, p, 0.254829592f);
  p = p * t;
  float er = 1.0f - p * __expf(-az * az);
  er = copysignf(er, z);
  return 0.5f * v * (1.0f + er);
}

// 16x32 bf16 fragment (A or B role) from LDS per the CDNA5 16-bit layout:
// lane L -> row (L&15); K = {base..base+7} U {base+16..base+23}, base = (L<16 ? 0 : 8).
__device__ __forceinline__ v16bf load_frag(const unsigned short* s, int row, int k0, int stride) {
  const int lane = threadIdx.x & 31;
  const int r  = row + (lane & 15);
  const int kb = k0 + ((lane >> 4) << 3);
  union { v16bf v; v4u u[2]; } f;
  f.u[0] = *reinterpret_cast<const v4u*>(s + r * stride + kb);
  f.u[1] = *reinterpret_cast<const v4u*>(s + r * stride + kb + 16);
  return f.v;
}

// ---- software-pipelined weight staging: global->VGPR, then convert->LDS ----
// One stage = 128(n) x 64(k) fp32 chunk = 8 float4 per thread (32 VGPRs).
struct WStage { v4f v[8]; };

__device__ __forceinline__ WStage stage_load(const float* __restrict__ w, int e, int ncol0, int kb) {
  const int tid = threadIdx.x;
  const float* src = w + ((size_t)e * D_DIM + ncol0) * (size_t)D_DIM + (size_t)kb * KCH;
  WStage s;
#pragma unroll
  for (int i = 0; i < 8; ++i) {
    int idx = tid + i * 256;          // 0..2047 over 128(n) x 16(float4)
    int r   = idx >> 4;
    int c4  = (idx & 15) << 2;
    s.v[i] = *reinterpret_cast<const v4f*>(src + (size_t)r * D_DIM + c4);
  }
  return s;
}

__device__ __forceinline__ void stage_store(const WStage& s, unsigned short* buf) {
  const int tid = threadIdx.x;
#pragma unroll
  for (int i = 0; i < 8; ++i) {
    int idx = tid + i * 256;
    int r   = idx >> 4;
    int c4  = (idx & 15) << 2;
    v2u p;
    p.x = pk2bf(s.v[i].x, s.v[i].y);
    p.y = pk2bf(s.v[i].z, s.v[i].w);
    *reinterpret_cast<v2u*>(buf + r * WB_STR + c4) = p;
  }
}

#define WMMA_BF16(A, B, C) \
  __builtin_amdgcn_wmma_f32_16x16x32_bf16(false, (A), false, (B), (short)0, (C), false, false)

__global__ __launch_bounds__(256)
void moe_lin_mlp_wmma(const float* __restrict__ x,
                      const float* __restrict__ w_lin, const float* __restrict__ b_lin,
                      const float* __restrict__ w1,    const float* __restrict__ b1,
                      const float* __restrict__ w2,    const float* __restrict__ b2,
                      float* __restrict__ out)
{
  __shared__ alignas(16) unsigned short xs[MT_ROWS * XS_STR];   // x tile, bf16    (~97 KB)
  __shared__ alignas(16) unsigned short hs[MT_ROWS * XS_STR];   // gelu tile, bf16 (~97 KB)
  __shared__ alignas(16) unsigned short wst[4 * WBUF];          // weight stages    (72 KB)

  // e-group-of-16 innermost (adjacent blocks share output cachelines; E is the
  // contiguous dim), m-tile next, e-group outer (weight window fits 192 MB L2).
  const int bid = blockIdx.x;
  const int e   = (bid >> 8) * 16 + (bid & 15);
  const int mt  = (bid >> 4) & 15;
  if (e >= E_NUM) return;

  const int tid  = threadIdx.x;
  const int wave = tid >> 5;
  const int lane = tid & 31;
  const int row0 = mt * MT_ROWS;

  // ---- load x tile (64 x 768 fp32) -> bf16 LDS ----
#pragma unroll
  for (int i = 0; i < 48; ++i) {
    int idx = tid + i * 256;
    int r   = idx / 192;
    int c4  = (idx - r * 192) << 2;
    v4f vv = *reinterpret_cast<const v4f*>(x + (size_t)(row0 + r) * D_DIM + c4);
    v2u p;
    p.x = pk2bf(vv.x, vv.y);
    p.y = pk2bf(vv.z, vv.w);
    *reinterpret_cast<v2u*>(xs + r * XS_STR + c4) = p;
  }
  __syncthreads();

  // ---- phase 1: h = gelu(x @ W1^T + b1), streamed over 6 N-chunks ----
  for (int nc = 0; nc < NNC; ++nc) {
    const int ncol0 = nc * NCHUNK;
    const int ncolL = ncol0 + (wave << 4) + (lane & 15);
    const float bias = b1[(size_t)e * D_DIM + ncolL];   // hoisted: completes under GEMM
    const v8f z = {0,0,0,0,0,0,0,0};
    v8f acc[NRT];
#pragma unroll
    for (int rt = 0; rt < NRT; ++rt) acc[rt] = z;

    stage_store(stage_load(w1, e, ncol0, 0), wst);
    __syncthreads();
    for (int kb = 0; kb < NKB; ++kb) {
      const unsigned short* cur = wst + (kb & 1) * WBUF;
      const bool have = (kb + 1 < NKB);
      WStage nxt;
      if (have) nxt = stage_load(w1, e, ncol0, kb + 1);       // loads issued early
#pragma unroll
      for (int ks = 0; ks < KCH; ks += 32) {
        v16bf bf = load_frag(cur, wave << 4, ks, WB_STR);
#pragma unroll
        for (int rt = 0; rt < NRT; ++rt) {
          v16bf a = load_frag(xs, rt << 4, kb * KCH + ks, XS_STR);
          acc[rt] = WMMA_BF16(a, bf, acc[rt]);
        }
      }
      if (have) stage_store(nxt, wst + ((kb + 1) & 1) * WBUF); // convert after WMMAs
      __syncthreads();
    }
    const int mrow = (lane >> 4) << 3;
#pragma unroll
    for (int rt = 0; rt < NRT; ++rt)
#pragma unroll
      for (int i = 0; i < 8; ++i)
        hs[((rt << 4) + mrow + i) * XS_STR + ncolL] = f2bf1(gelu_exact(acc[rt][i] + bias));
    __syncthreads();
  }

  // ---- phase 2: out = x @ W_lin^T + h @ W2^T + (b_lin + b2), fused per N-chunk ----
  for (int nc = 0; nc < NNC; ++nc) {
    const int ncol0 = nc * NCHUNK;
    const int ncolL = ncol0 + (wave << 4) + (lane & 15);
    const float bsum = b_lin[(size_t)e * D_DIM + ncolL] + b2[(size_t)e * D_DIM + ncolL];
    const v8f z = {0,0,0,0,0,0,0,0};
    v8f aL[NRT], aM[NRT];
#pragma unroll
    for (int rt = 0; rt < NRT; ++rt) { aL[rt] = z; aM[rt] = z; }

    stage_store(stage_load(w_lin, e, ncol0, 0), wst);
    stage_store(stage_load(w2,    e, ncol0, 0), wst + 2 * WBUF);
    __syncthreads();
    for (int kb = 0; kb < NKB; ++kb) {
      const unsigned short* curL = wst + (kb & 1) * WBUF;
      const unsigned short* curM = wst + (2 + (kb & 1)) * WBUF;
      const bool have = (kb + 1 < NKB);
      WStage nL, nM;
      if (have) nL = stage_load(w_lin, e, ncol0, kb + 1);
#pragma unroll
      for (int ks = 0; ks < KCH; ks += 32) {                  // lin branch
        v16bf bL = load_frag(curL, wave << 4, ks, WB_STR);
#pragma unroll
        for (int rt = 0; rt < NRT; ++rt) {
          v16bf xf = load_frag(xs, rt << 4, kb * KCH + ks, XS_STR);
          aL[rt] = WMMA_BF16(xf, bL, aL[rt]);
        }
      }
      if (have) { stage_store(nL, wst + ((kb + 1) & 1) * WBUF);
                  nM = stage_load(w2, e, ncol0, kb + 1); }
#pragma unroll
      for (int ks = 0; ks < KCH; ks += 32) {                  // mlp branch
        v16bf bM = load_frag(curM, wave << 4, ks, WB_STR);
#pragma unroll
        for (int rt = 0; rt < NRT; ++rt) {
          v16bf hf = load_frag(hs, rt << 4, kb * KCH + ks, XS_STR);
          aM[rt] = WMMA_BF16(hf, bM, aM[rt]);
        }
      }
      if (have) stage_store(nM, wst + (2 + ((kb + 1) & 1)) * WBUF);
      __syncthreads();
    }
    const int mrow = (lane >> 4) << 3;
#pragma unroll
    for (int rt = 0; rt < NRT; ++rt)
#pragma unroll
      for (int i = 0; i < 8; ++i) {
        size_t r = (size_t)(row0 + (rt << 4) + mrow + i);
        out[(r * D_DIM + ncolL) * E_NUM + e] = aL[rt][i] + aM[rt][i] + bsum;
      }
    __syncthreads();
  }
}

extern "C" void kernel_launch(void* const* d_in, const int* in_sizes, int n_in,
                              void* d_out, int out_size, void* d_ws, size_t ws_size,
                              hipStream_t stream) {
  (void)in_sizes; (void)n_in; (void)out_size; (void)d_ws; (void)ws_size;
  const float* x     = (const float*)d_in[0];
  const float* w_lin = (const float*)d_in[1];
  const float* b_lin = (const float*)d_in[2];
  const float* w1    = (const float*)d_in[3];
  const float* b1    = (const float*)d_in[4];
  const float* w2    = (const float*)d_in[5];
  const float* b2    = (const float*)d_in[6];
  float* out = (float*)d_out;
  // 7 e-groups of 16 (last partial, guarded) x 16 m-tiles x 16 e-in-group = 1792 blocks
  moe_lin_mlp_wmma<<<dim3(7 * 16 * 16), dim3(256), 0, stream>>>(
      x, w_lin, b_lin, w1, b1, w2, b2, out);
}
